// Disc_edge_15573551415682
// MI455X (gfx1250) — matrix operand order repaired and checked by reading, not compiled, not run on hardware
//
#include <hip/hip_runtime.h>
#include <hip/hip_bf16.h>
#include <cstdint>

// ---------------------------------------------------------------------------
// GNN discriminator for MI455X (gfx1250, wave32, WMMA).
//
// Roofline: per edge-layer traffic = 134MB read + 134MB write of e (f32).
// At 23.3 TB/s this dominates; the per-edge GEMM is factorized so that only
// the e-dependent 64-wide K chunk runs through v_wmma_f32_16x16x32_f16
// (f16 inputs, f32 accumulation). x_i and x_j contributions are rank-deficient
// and precomputed as per-row vectors (c0, G), added in the epilogue.
// Layer 2 never stores e3: only the per-batch mean survives.
// ---------------------------------------------------------------------------

typedef __attribute__((ext_vector_type(16))) _Float16 v16h;
typedef __attribute__((ext_vector_type(8)))  _Float16 v8h;
typedef __attribute__((ext_vector_type(8)))  float    v8f;

union F16x16 { v16h v; v8h h[2]; };

#define GNN_B  8
#define GNN_N  256
#define GNN_F  64
#define GNN_BN (GNN_B * GNN_N)

// ---------------------------------------------------------------------------
// Per-layer precompute:  c0[row,f] = x[row,:] @ We[0:64, f] + be[f]
//                        G [row,f] = x[row,:] @ We[64:128, f]
// grid = B*N blocks x 64 threads. Tiny (~34 MFLOP per launch); VALU is fine.
// ---------------------------------------------------------------------------
__global__ __launch_bounds__(64)
void pre_kernel(const float* __restrict__ x, const float* __restrict__ We,
                const float* __restrict__ be,
                float* __restrict__ c0, float* __restrict__ G)
{
    __shared__ float xr[64];
    const int row = blockIdx.x;
    const int f   = threadIdx.x;
    xr[f] = x[(size_t)row * 64 + f];
    __syncthreads();
    float s0 = be[f];
    float s1 = 0.0f;
#pragma unroll 8
    for (int k = 0; k < 64; ++k) {
        float xv = xr[k];
        s0 += xv * We[k * 64 + f];          // xi part: We rows 0..63
        s1 += xv * We[(64 + k) * 64 + f];   // xj part: We rows 64..127
    }
    c0[(size_t)row * 64 + f] = s0;
    G [(size_t)row * 64 + f] = s1;
}

// ---------------------------------------------------------------------------
// Edge update kernel. One workgroup per (b, i): a 256x64 @ 64x64 GEMM on the
// e-chunk via WMMA, fused epilogue: + c0[b,i,:] + G[b,j,:], relu, * A[b,i,j].
// MODE 0: store e_new + write agg = colsum/deg      (layer 0)
// MODE 1: store e_new only (in-place safe: tile staged in LDS)  (layer 1)
// MODE 2: no store; write colsum partials for edge-mean        (layer 2)
// ---------------------------------------------------------------------------
template<int MODE>
__global__ __launch_bounds__(256)
void edge_kernel(const int*   __restrict__ Aidx,   // [B,N,N] 0/1
                 const float* __restrict__ e_in,   // [B,N,N,64]
                 float*       __restrict__ e_out,  // [B,N,N,64] (MODE<2)
                 const float* __restrict__ c0g,    // [B*N,64]
                 const float* __restrict__ Gg,     // [B*N,64]
                 const float* __restrict__ Wee,    // We rows 128..191, [64][64]
                 float*       __restrict__ red_out)// [B*N,64] agg or partial
{
    // LDS: e tile as f16 (rows padded to 72 halfs = 144B, 16B aligned,
    // spreads banks for ds_load_b128 fragment gathers), We^T as f16.
    __shared__ _Float16 lds_e [256][72];   // 36 KB  [j][k]
    __shared__ _Float16 lds_wT[ 64][72];   //  9 KB  [n][k]  (We_e transposed)
    __shared__ float    lds_A [256];       //  1 KB
    __shared__ float    lds_part[8][64];   //  2 KB  per-wave column sums

    const int tid = threadIdx.x;
    const int bi  = blockIdx.x;        // b*256 + i
    const int b   = bi >> 8;

    // ---- stage e[b,i,:,:] -> f16 LDS (thread t handles row j=t, 16x float4)
    {
        const float4* src = (const float4*)(e_in + ((size_t)bi * 256 + tid) * 64);
#pragma unroll
        for (int q = 0; q < 16; ++q) {
            float4 v = src[q];
            lds_e[tid][q * 4 + 0] = (_Float16)v.x;
            lds_e[tid][q * 4 + 1] = (_Float16)v.y;
            lds_e[tid][q * 4 + 2] = (_Float16)v.z;
            lds_e[tid][q * 4 + 3] = (_Float16)v.w;
        }
    }
    // ---- stage We_e^T: lds_wT[n][k] = Wee[k*64+n]  (4096 elems, L2-resident)
    {
#pragma unroll
        for (int q = 0; q < 16; ++q) {
            int flat = tid * 16 + q;
            int k = flat >> 6;
            int n = flat & 63;
            lds_wT[n][k] = (_Float16)Wee[k * 64 + n];
        }
    }
    lds_A[tid] = (float)Aidx[(size_t)bi * 256 + tid];
    __syncthreads();

    const int wv   = tid >> 5;       // wave 0..7 -> rows [wv*32, wv*32+32)
    const int lane = tid & 31;
    const int lg   = lane >> 4;      // lane group (0: lanes 0-15, 1: 16-31)
    const int lr   = lane & 15;
    const int m0   = wv * 32;

    // ---- A fragments (16x32 f16): elem e -> K = (e/8)*16 + lg*8 + e%8
    F16x16 af[2][2];
#pragma unroll
    for (int mt = 0; mt < 2; ++mt)
#pragma unroll
        for (int ks = 0; ks < 2; ++ks) {
            int row = m0 + mt * 16 + lr;
            int kb  = ks * 32 + lg * 8;
            af[mt][ks].h[0] = *(const v8h*)&lds_e[row][kb];
            af[mt][ks].h[1] = *(const v8h*)&lds_e[row][kb + 16];
        }
    // ---- B fragments (32x16 f16): lane holds column n=lr, K = (e/8)*16 + lg*8 + e%8
    F16x16 bf[4][2];
#pragma unroll
    for (int nt = 0; nt < 4; ++nt)
#pragma unroll
        for (int ks = 0; ks < 2; ++ks) {
            int col = nt * 16 + lr;
            int kb  = ks * 32 + lg * 8;
            bf[nt][ks].h[0] = *(const v8h*)&lds_wT[col][kb];
            bf[nt][ks].h[1] = *(const v8h*)&lds_wT[col][kb + 16];
        }

    // ---- 16 WMMAs per wave, epilogue fused per 16x16 tile
    float cs[4] = {0.f, 0.f, 0.f, 0.f};
#pragma unroll
    for (int nt = 0; nt < 4; ++nt) {
        const int   n   = nt * 16 + lr;
        const float c0v = c0g[(size_t)bi * 64 + n];
#pragma unroll
        for (int mt = 0; mt < 2; ++mt) {
            v8f acc = {};
            acc = __builtin_amdgcn_wmma_f32_16x16x32_f16(
                false, af[mt][0].v, false, bf[nt][0].v, (short)0, acc, false, false);
            acc = __builtin_amdgcn_wmma_f32_16x16x32_f16(
                false, af[mt][1].v, false, bf[nt][1].v, (short)0, acc, false, false);
            // C/D layout: VGPR r, lane group lg -> M = lg*8 + r, N = lr
#pragma unroll
            for (int r = 0; r < 8; ++r) {
                int   j   = m0 + mt * 16 + lg * 8 + r;
                float g   = Gg[((size_t)b * 256 + j) * 64 + n];
                float val = acc[r] + c0v + g;
                val = fmaxf(val, 0.0f) * lds_A[j];
                if (MODE < 2)
                    e_out[((size_t)bi * 256 + j) * 64 + n] = val;
                if (MODE != 1)
                    cs[nt] += val;
            }
        }
    }

    // ---- deterministic column-sum reduction (no float atomics)
    if (MODE != 1) {
#pragma unroll
        for (int nt = 0; nt < 4; ++nt) {
            float tot = cs[nt] + __shfl_xor(cs[nt], 16, 32); // merge lane groups
            if (lg == 0) lds_part[wv][nt * 16 + lr] = tot;
        }
        __syncthreads();
        if (tid < 64) {
            float tot = 0.0f;
#pragma unroll
            for (int w = 0; w < 8; ++w) tot += lds_part[w][tid];
            if (MODE == 0) {
                float deg = 0.0f;
                for (int j = 0; j < 256; ++j) deg += lds_A[j];
                deg = fmaxf(deg, 1.0f);
                red_out[(size_t)bi * 64 + tid] = tot / deg;
            } else {
                red_out[(size_t)bi * 64 + tid] = tot;   // raw colsum for mean
            }
        }
    }
}

// ---------------------------------------------------------------------------
// Node update (layer 0 only): x1 = relu([x | agg] @ Wn + bn)
// ---------------------------------------------------------------------------
__global__ __launch_bounds__(64)
void node_kernel(const float* __restrict__ x, const float* __restrict__ agg,
                 const float* __restrict__ Wn, const float* __restrict__ bn,
                 float* __restrict__ x1)
{
    __shared__ float h[128];
    const int row = blockIdx.x;
    const int f   = threadIdx.x;
    h[f]      = x  [(size_t)row * 64 + f];
    h[64 + f] = agg[(size_t)row * 64 + f];
    __syncthreads();
    float s = bn[f];
#pragma unroll 8
    for (int k = 0; k < 128; ++k) s += h[k] * Wn[k * 64 + f];
    x1[(size_t)row * 64 + f] = fmaxf(s, 0.0f);
}

// ---------------------------------------------------------------------------
// Head: per-batch mean of layer-2 partials -> 64->64->64->1 MLP
// ---------------------------------------------------------------------------
__global__ __launch_bounds__(64)
void head_kernel(const float* __restrict__ partial,
                 const float* __restrict__ W1, const float* __restrict__ b1,
                 const float* __restrict__ W2, const float* __restrict__ b2,
                 const float* __restrict__ W3, const float* __restrict__ b3,
                 float* __restrict__ out)
{
    __shared__ float v[64], h1[64], h2[64];
    const int b = blockIdx.x;
    const int f = threadIdx.x;
    float s = 0.0f;
    for (int i = 0; i < 256; ++i) s += partial[((size_t)b * 256 + i) * 64 + f];
    v[f] = s * (1.0f / 65536.0f);   // mean over N*N
    __syncthreads();
    s = b1[f];
    for (int k = 0; k < 64; ++k) s += v[k] * W1[k * 64 + f];
    h1[f] = fmaxf(s, 0.0f);
    __syncthreads();
    s = b2[f];
    for (int k = 0; k < 64; ++k) s += h1[k] * W2[k * 64 + f];
    h2[f] = fmaxf(s, 0.0f);
    __syncthreads();
    if (f == 0) {
        float o = b3[0];
        for (int k = 0; k < 64; ++k) o += h2[k] * W3[k];
        out[b] = o;
    }
}

// ---------------------------------------------------------------------------
extern "C" void kernel_launch(void* const* d_in, const int* in_sizes, int n_in,
                              void* d_out, int out_size, void* d_ws, size_t ws_size,
                              hipStream_t stream)
{
    (void)in_sizes; (void)n_in; (void)out_size; (void)ws_size;

    const int*   Aidx  = (const int*)  d_in[0];
    const float* x     = (const float*)d_in[1];
    const float* eattr = (const float*)d_in[2];
    const float* We0 = (const float*)d_in[3],  *be0 = (const float*)d_in[4];
    const float* Wn0 = (const float*)d_in[5],  *bn0 = (const float*)d_in[6];
    const float* We1 = (const float*)d_in[7],  *be1 = (const float*)d_in[8];
    const float* We2 = (const float*)d_in[11], *be2 = (const float*)d_in[12];
    const float* W1  = (const float*)d_in[15], *b1  = (const float*)d_in[16];
    const float* W2  = (const float*)d_in[17], *b2  = (const float*)d_in[18];
    const float* W3  = (const float*)d_in[19], *b3  = (const float*)d_in[20];
    float* out = (float*)d_out;

    // workspace layout (all 256B aligned)
    char*  ws    = (char*)d_ws;
    size_t off   = 0;
    float* ws_e  = (float*)(ws + off); off += (size_t)GNN_B * GNN_N * GNN_N * GNN_F * 4; // 134MB
    float* c0    = (float*)(ws + off); off += (size_t)GNN_BN * GNN_F * 4;
    float* G     = (float*)(ws + off); off += (size_t)GNN_BN * GNN_F * 4;
    float* agg   = (float*)(ws + off); off += (size_t)GNN_BN * GNN_F * 4;
    float* x1    = (float*)(ws + off); off += (size_t)GNN_BN * GNN_F * 4;
    float* part  = (float*)(ws + off); off += (size_t)GNN_BN * GNN_F * 4;

    // ---- layer 0: edges (store + agg) then node update
    pre_kernel<<<GNN_BN, 64, 0, stream>>>(x, We0, be0, c0, G);
    edge_kernel<0><<<GNN_BN, 256, 0, stream>>>(Aidx, eattr, ws_e, c0, G,
                                               We0 + 128 * 64, agg);
    node_kernel<<<GNN_BN, 64, 0, stream>>>(x, agg, Wn0, bn0, x1);

    // ---- layer 1: edges only, in-place on ws_e (tile fully staged in LDS)
    pre_kernel<<<GNN_BN, 64, 0, stream>>>(x1, We1, be1, c0, G);
    edge_kernel<1><<<GNN_BN, 256, 0, stream>>>(Aidx, ws_e, ws_e, c0, G,
                                               We1 + 128 * 64, nullptr);

    // ---- layer 2: no e3 store, only per-(b,i) column sums for the mean
    pre_kernel<<<GNN_BN, 64, 0, stream>>>(x1, We2, be2, c0, G);
    edge_kernel<2><<<GNN_BN, 256, 0, stream>>>(Aidx, ws_e, nullptr, c0, G,
                                               We2 + 128 * 64, part);

    // ---- edge-mean + MLP head
    head_kernel<<<GNN_B, 64, 0, stream>>>(part, W1, b1, W2, b2, W3, b3, out);
}
